// Sggnn_23218593202512
// MI455X (gfx1250) — matrix operand order repaired and checked by reading, not compiled, not run on hardware
//
#include <hip/hip_runtime.h>

typedef __attribute__((ext_vector_type(16))) __bf16 v16bf;
typedef __attribute__((ext_vector_type(8)))  __bf16 v8bf;
typedef __attribute__((ext_vector_type(8)))  float  v8f;
typedef __attribute__((ext_vector_type(4)))  float  v4f;
typedef __attribute__((ext_vector_type(8)))  unsigned short u16x8;
typedef __attribute__((ext_vector_type(4)))  unsigned short u16x4;

#define PP 128
#define GG 256
#define FF 1024
#define HH 512

static __device__ __forceinline__ unsigned short f2bfbits(float f) {
  union { float f; unsigned u; } x; x.f = f;
  unsigned r = x.u + 0x7FFFu + ((x.u >> 16) & 1u);   // round-to-nearest-even
  return (unsigned short)(r >> 16);
}
static __device__ __forceinline__ float bfbits2f(unsigned short h) {
  union { unsigned u; float f; } x; x.u = ((unsigned)h) << 16; return x.f;
}

// ---------------- fold BN (eval) into per-channel affine: y = x*s + t ----------------
__global__ void k_affine(const float* __restrict__ g, const float* __restrict__ b,
                         const float* __restrict__ m, const float* __restrict__ v,
                         float* __restrict__ s, float* __restrict__ t, int n) {
  int i = blockIdx.x * blockDim.x + threadIdx.x;
  if (i < n) {
    float sc = g[i] * rsqrtf(v[i] + 1e-5f);
    s[i] = sc;
    t[i] = b[i] - m[i] * sc;
  }
}

// ------ d[i,j,f] = bf16( ((X[i,f]-Y[j,f])^2) * s[f] + t[f] ), 4 chans/thread ------
__global__ void __launch_bounds__(256) k_pairdiff(const float* __restrict__ X, const float* __restrict__ Y,
                                                  const float* __restrict__ s, const float* __restrict__ t,
                                                  unsigned short* __restrict__ out, int Gdim) {
  int pair = blockIdx.x;
  int i = pair / Gdim, j = pair % Gdim;
  int f4 = threadIdx.x;                       // 256 threads * 4 = FF
  v4f x  = reinterpret_cast<const v4f*>(X + (size_t)i * FF)[f4];
  v4f y  = reinterpret_cast<const v4f*>(Y + (size_t)j * FF)[f4];
  v4f ss = reinterpret_cast<const v4f*>(s)[f4];
  v4f tt = reinterpret_cast<const v4f*>(t)[f4];
  u16x4 o;
#pragma unroll
  for (int e = 0; e < 4; ++e) {
    float d = x[e] - y[e];
    o[e] = f2bfbits(d * d * ss[e] + tt[e]);
  }
  reinterpret_cast<u16x4*>(out + (size_t)pair * FF)[f4] = o;
}

// -------- pack W[K,N] (f32, row-major) into WMMA B-fragment order (bf16) --------
// Wp[((nt*(K/32)+kt)*32 + lane)*16 + e] = W[kt*32 + 16*(lane>>4) + e, nt*16 + (lane&15)]
__global__ void k_repack(const float* __restrict__ W, unsigned short* __restrict__ Wp, int K, int N) {
  size_t idx = (size_t)blockIdx.x * blockDim.x + threadIdx.x;   // exactly K*N threads
  int e = (int)(idx & 15);
  int L = (int)((idx >> 4) & 31);
  size_t tile = idx >> 9;
  int ktiles = K >> 5;
  int kt = (int)(tile % ktiles);
  int nt = (int)(tile / ktiles);
  int k = kt * 32 + ((L >> 4) << 4) + e;
  int n = (nt << 4) + (L & 15);
  Wp[idx] = f2bfbits(W[(size_t)k * N + n]);
}

// -------- GEMM: out = lrelu( (A[M,K] @ W + bias) * s + t ) in bf16 --------
// 256 threads = 8 waves; wave -> 16(M) x 64(N) tile; A bf16 row-major, W pre-packed.
// packB==0: row-major output.  packB==1: output written per-256-row slab in
// WMMA B-fragment order (for consumption as the B operand of the diffusion GEMM).
__global__ void __launch_bounds__(256) k_gemm(const unsigned short* __restrict__ A,
                                              const unsigned short* __restrict__ Wp,
                                              const float* __restrict__ bias,
                                              const float* __restrict__ s,
                                              const float* __restrict__ t,
                                              unsigned short* __restrict__ out,
                                              int M, int N, int K, int packB) {
  const int lane = threadIdx.x & 31;
  const int wave = threadIdx.x >> 5;
  const int hi = lane >> 4;
  const int lm = lane & 15;
  const int m0 = blockIdx.y * 128 + wave * 16;
  const int n0 = blockIdx.x * 64;
  const int ktiles = K >> 5;

  const unsigned short* arow = A + (size_t)(m0 + lm) * K + 16 * hi;
  const v16bf* wp = reinterpret_cast<const v16bf*>(Wp);
  const size_t wbase = (size_t)(n0 >> 4) * ktiles * 32 + lane;

  v8f zero = {};
  v8f acc[4];
#pragma unroll
  for (int j = 0; j < 4; ++j) acc[j] = zero;

  for (int kt = 0; kt < ktiles; ++kt) {
    v8bf alo = *reinterpret_cast<const v8bf*>(arow + kt * 32);
    v8bf ah2 = *reinterpret_cast<const v8bf*>(arow + kt * 32 + 8);
    v16bf af = __builtin_shufflevector(alo, ah2, 0,1,2,3,4,5,6,7,8,9,10,11,12,13,14,15);
#pragma unroll
    for (int j = 0; j < 4; ++j) {
      v16bf bf = wp[wbase + ((size_t)j * ktiles + kt) * 32];
      acc[j] = __builtin_amdgcn_wmma_f32_16x16x32_bf16(false, af, false, bf,
                                                       (short)0, acc[j], false, false);
    }
  }

#pragma unroll
  for (int j = 0; j < 4; ++j) {
    int n = n0 + j * 16 + lm;
    float sc = s[n], sh = t[n], bi = bias[n];
    u16x8 res;
#pragma unroll
    for (int v = 0; v < 8; ++v) {
      float val = acc[j][v] + bi;
      val = val * sc + sh;
      val = val > 0.f ? val : 0.1f * val;
      res[v] = f2bfbits(val);
    }
    if (packB) {
      // rows m = m0 + 8*hi + v  ->  g = (m0&255) + 8*hi + v inside slab p = m0>>8
      int p  = m0 >> 8;
      int gb = (m0 & 255) + 8 * hi;                 // v runs over gb..gb+7, same 16-block
      size_t base = ((size_t)(n >> 4) * (GG >> 5) + (gb >> 5)) * 512
                  + (size_t)((((gb >> 4) & 1) << 4) + (n & 15)) * 16
                  + (gb & 15);
      *reinterpret_cast<u16x8*>(out + (size_t)p * (GG * FF) + base) = res;
    } else {
#pragma unroll
      for (int v = 0; v < 8; ++v)
        out[(size_t)(m0 + hi * 8 + v) * N + n] = res[v];
    }
  }
}

// -------- graph diffusion: for each p, out_p[h,f] = sum_g wT[h,g] * Tp[p][g,f] --------
// A = wT (bf16 row-major 256x256); B = t-slab pre-packed in fragment order by k_gemm.
__global__ void __launch_bounds__(256) k_einsum(const unsigned short* __restrict__ wT,
                                                const unsigned short* __restrict__ Tp,
                                                unsigned short* __restrict__ out) {
  const int lane = threadIdx.x & 31;
  const int wave = threadIdx.x >> 5;
  const int hi = lane >> 4;
  const int lm = lane & 15;
  const int p  = blockIdx.z;
  const int m0 = blockIdx.y * 128 + wave * 16;   // h
  const int n0 = blockIdx.x * 64;                // f
  const int ktiles = GG >> 5;                    // 8

  const unsigned short* arow = wT + (size_t)(m0 + lm) * GG + 16 * hi;
  const v16bf* wp = reinterpret_cast<const v16bf*>(Tp + (size_t)p * (GG * FF));
  const size_t wbase = (size_t)(n0 >> 4) * ktiles * 32 + lane;

  v8f zero = {};
  v8f acc[4];
#pragma unroll
  for (int j = 0; j < 4; ++j) acc[j] = zero;

  for (int kt = 0; kt < ktiles; ++kt) {
    v8bf alo = *reinterpret_cast<const v8bf*>(arow + kt * 32);
    v8bf ah2 = *reinterpret_cast<const v8bf*>(arow + kt * 32 + 8);
    v16bf af = __builtin_shufflevector(alo, ah2, 0,1,2,3,4,5,6,7,8,9,10,11,12,13,14,15);
#pragma unroll
    for (int j = 0; j < 4; ++j) {
      v16bf bf = wp[wbase + ((size_t)j * ktiles + kt) * 32];
      acc[j] = __builtin_amdgcn_wmma_f32_16x16x32_bf16(false, af, false, bf,
                                                       (short)0, acc[j], false, false);
    }
  }

#pragma unroll
  for (int j = 0; j < 4; ++j) {
    int n = n0 + j * 16 + lm;
#pragma unroll
    for (int v = 0; v < 8; ++v) {
      int m = m0 + hi * 8 + v;
      out[((size_t)p * GG + m) * FF + n] = f2bfbits(acc[j][v]);
    }
  }
}

// -------- row-dot producing the transposed affinity matrix wT[h*G+g] = w[g,h] --------
__global__ void __launch_bounds__(256) k_rowdot_w(const unsigned short* __restrict__ Xr,
                                                  const float* __restrict__ wv,
                                                  const float* __restrict__ b,
                                                  unsigned short* __restrict__ wT) {
  int row = blockIdx.x * 8 + (threadIdx.x >> 5);  // row = g1*G + g2
  int lane = threadIdx.x & 31;
  const unsigned short* xr = Xr + (size_t)row * HH;
  float acc = 0.f;
#pragma unroll
  for (int i = 0; i < HH / 32; ++i) acc += bfbits2f(xr[lane + 32 * i]) * wv[lane + 32 * i];
#pragma unroll
  for (int off = 16; off > 0; off >>= 1) acc += __shfl_down(acc, off, 32);
  if (lane == 0) {
    int g1 = row >> 8, g2 = row & 255;
    wT[(size_t)g2 * GG + g1] = f2bfbits(acc + b[0]);   // store transposed
  }
}

// -------- final row-dot: out[r] = feat[r,:] . cls_W + cls_b  (f32 output) --------
__global__ void __launch_bounds__(256) k_rowdot_out(const unsigned short* __restrict__ Xr,
                                                    const float* __restrict__ wv,
                                                    const float* __restrict__ b,
                                                    float* __restrict__ out) {
  int row = blockIdx.x * 8 + (threadIdx.x >> 5);
  int lane = threadIdx.x & 31;
  const unsigned short* xr = Xr + (size_t)row * HH;
  float acc = 0.f;
#pragma unroll
  for (int i = 0; i < HH / 32; ++i) acc += bfbits2f(xr[lane + 32 * i]) * wv[lane + 32 * i];
#pragma unroll
  for (int off = 16; off > 0; off >>= 1) acc += __shfl_down(acc, off, 32);
  if (lane == 0) out[row] = acc + b[0];
}

extern "C" void kernel_launch(void* const* d_in, const int* in_sizes, int n_in,
                              void* d_out, int out_size, void* d_ws, size_t ws_size,
                              hipStream_t stream) {
  (void)in_sizes; (void)n_in; (void)out_size; (void)ws_size;
  const float* f_p   = (const float*)d_in[0];
  const float* f_g   = (const float*)d_in[1];
  const float* bn_g  = (const float*)d_in[2];
  const float* bn_b  = (const float*)d_in[3];
  const float* bn_m  = (const float*)d_in[4];
  const float* bn_v  = (const float*)d_in[5];
  const float* rf_W1 = (const float*)d_in[6];
  const float* rf_b1 = (const float*)d_in[7];
  const float* rf1_g = (const float*)d_in[8];
  const float* rf1_b = (const float*)d_in[9];
  const float* rf1_m = (const float*)d_in[10];
  const float* rf1_v = (const float*)d_in[11];
  const float* rf_W2 = (const float*)d_in[12];
  const float* rf_b2 = (const float*)d_in[13];
  const float* rf2_g = (const float*)d_in[14];
  const float* rf2_b = (const float*)d_in[15];
  const float* rf2_m = (const float*)d_in[16];
  const float* rf2_v = (const float*)d_in[17];
  const float* sfc_W = (const float*)d_in[18];
  const float* sfc_b = (const float*)d_in[19];
  const float* sbn_g = (const float*)d_in[20];
  const float* sbn_b = (const float*)d_in[21];
  const float* sbn_m = (const float*)d_in[22];
  const float* sbn_v = (const float*)d_in[23];
  const float* scl_W = (const float*)d_in[24];
  const float* scl_b = (const float*)d_in[25];
  const float* ffc_W = (const float*)d_in[26];
  const float* ffc_b = (const float*)d_in[27];
  const float* fbn_g = (const float*)d_in[28];
  const float* fbn_b = (const float*)d_in[29];
  const float* fbn_m = (const float*)d_in[30];
  const float* fbn_v = (const float*)d_in[31];
  const float* cls_W = (const float*)d_in[32];
  const float* cls_b = (const float*)d_in[33];

  const size_t MB = 1ull << 20;
  char* ws = (char*)d_ws;
  // big buffers (reused across phases; live ranges do not overlap)
  unsigned short* bufA  = (unsigned short*)(ws);             // [0,128MB): dg ; later d, feat
  unsigned short* bufA2 = (unsigned short*)(ws + 64 * MB);   // [64,128MB): h ; later d_new
  unsigned short* bufB  = (unsigned short*)(ws + 128 * MB);  // [128,192MB): hg ; later t (packed)
  char* wp = ws + 192 * MB;
  unsigned short* rf_W1p = (unsigned short*)wp; wp += 2 * MB;
  unsigned short* rf_W2p = (unsigned short*)wp; wp += 2 * MB;
  unsigned short* sfc_Wp = (unsigned short*)wp; wp += 1 * MB;
  unsigned short* ffc_Wp = (unsigned short*)wp; wp += 1 * MB;
  unsigned short* wT     = (unsigned short*)wp; wp += 128 * 1024;
  float* s_bn = (float*)wp; wp += FF * 4;
  float* t_bn = (float*)wp; wp += FF * 4;
  float* s_r1 = (float*)wp; wp += FF * 4;
  float* t_r1 = (float*)wp; wp += FF * 4;
  float* s_r2 = (float*)wp; wp += FF * 4;
  float* t_r2 = (float*)wp; wp += FF * 4;
  float* s_sb = (float*)wp; wp += HH * 4;
  float* t_sb = (float*)wp; wp += HH * 4;
  float* s_fb = (float*)wp; wp += HH * 4;
  float* t_fb = (float*)wp; wp += HH * 4;

  // 1) fold all BNs into affines + repack weights to bf16 fragment layout
  k_affine<<<dim3((FF + 255) / 256), dim3(256), 0, stream>>>(bn_g, bn_b, bn_m, bn_v, s_bn, t_bn, FF);
  k_affine<<<dim3((FF + 255) / 256), dim3(256), 0, stream>>>(rf1_g, rf1_b, rf1_m, rf1_v, s_r1, t_r1, FF);
  k_affine<<<dim3((FF + 255) / 256), dim3(256), 0, stream>>>(rf2_g, rf2_b, rf2_m, rf2_v, s_r2, t_r2, FF);
  k_affine<<<dim3((HH + 255) / 256), dim3(256), 0, stream>>>(sbn_g, sbn_b, sbn_m, sbn_v, s_sb, t_sb, HH);
  k_affine<<<dim3((HH + 255) / 256), dim3(256), 0, stream>>>(fbn_g, fbn_b, fbn_m, fbn_v, s_fb, t_fb, HH);
  k_repack<<<dim3((FF * FF) / 256), dim3(256), 0, stream>>>(rf_W1, rf_W1p, FF, FF);
  k_repack<<<dim3((FF * FF) / 256), dim3(256), 0, stream>>>(rf_W2, rf_W2p, FF, FF);
  k_repack<<<dim3((FF * HH) / 256), dim3(256), 0, stream>>>(sfc_W, sfc_Wp, FF, HH);
  k_repack<<<dim3((FF * HH) / 256), dim3(256), 0, stream>>>(ffc_W, ffc_Wp, FF, HH);

  // 2) gallery-gallery branch: dg -> hg -> w (stored transposed, bf16)
  k_pairdiff<<<dim3(GG * GG), dim3(256), 0, stream>>>(f_g, f_g, s_bn, t_bn, bufA, GG);
  k_gemm<<<dim3(HH / 64, (GG * GG) / 128), dim3(256), 0, stream>>>(
      bufA, sfc_Wp, sfc_b, s_sb, t_sb, bufB, GG * GG, HH, FF, 0);
  k_rowdot_w<<<dim3((GG * GG) / 8), dim3(256), 0, stream>>>(bufB, scl_W, scl_b, wT);

  // 3) probe-gallery branch: d -> h -> t (t written in packed B-fragment order)
  k_pairdiff<<<dim3(PP * GG), dim3(256), 0, stream>>>(f_p, f_g, s_bn, t_bn, bufA, GG);
  k_gemm<<<dim3(FF / 64, (PP * GG) / 128), dim3(256), 0, stream>>>(
      bufA, rf_W1p, rf_b1, s_r1, t_r1, bufA2, PP * GG, FF, FF, 0);
  k_gemm<<<dim3(FF / 64, (PP * GG) / 128), dim3(256), 0, stream>>>(
      bufA2, rf_W2p, rf_b2, s_r2, t_r2, bufB, PP * GG, FF, FF, 1);

  // 4) graph diffusion: d_new[p,h,f] = sum_g wT[h,g] * t[p,g,f]  (packed B loads)
  k_einsum<<<dim3(FF / 64, GG / 128, PP), dim3(256), 0, stream>>>(wT, bufB, bufA2);

  // 5) per-pair FC + classifier
  k_gemm<<<dim3(HH / 64, (PP * GG) / 128), dim3(256), 0, stream>>>(
      bufA2, ffc_Wp, ffc_b, s_fb, t_fb, bufA, PP * GG, HH, FF, 0);
  k_rowdot_out<<<dim3((PP * GG) / 8), dim3(256), 0, stream>>>(bufA, cls_W, cls_b, (float*)d_out);
}
